// MultiheadFlashDiff2_3942779978251
// MI455X (gfx1250) — matrix operand order repaired and checked by reading, not compiled, not run on hardware
//
#include <hip/hip_runtime.h>
#include <hip/hip_bf16.h>
#include <math.h>

// ---------------------------------------------------------------------------
// Problem constants (from reference)
// ---------------------------------------------------------------------------
#define S_LEN 2048
#define EMB   1024
#define NH    8        // value heads
#define NQH   16       // 2*NH query/key heads of dim 64
#define HD    64       // q/k head dim
#define VD    128      // v head dim
#define LAMBDA_INIT_F 0.7836057665316245f   // 0.8 - 0.6*exp(-0.3*12)
#define RMS_EPS_F     1e-5f

// ---------------------------------------------------------------------------
// WMMA types / helpers (CDNA5 gfx1250, wave32)
// ---------------------------------------------------------------------------
typedef __attribute__((ext_vector_type(16))) __bf16 v16bf;
typedef __attribute__((ext_vector_type(8)))  float  v8f;

union BF16Frag { v16bf v; unsigned short u[16]; uint4 q[2]; };

__device__ __forceinline__ unsigned short f2bf(float f) {
  union { float f; unsigned int i; } c; c.f = f;
  unsigned int r = c.i + 0x7FFFu + ((c.i >> 16) & 1u);   // round-to-nearest-even
  return (unsigned short)(r >> 16);
}

__device__ __forceinline__ v8f wmma_bf16(v16bf a, v16bf b, v8f c) {
  // 8 args: (neg_a, A, neg_b, B, c_mod, C, reuse_a, reuse_b)
  return __builtin_amdgcn_wmma_f32_16x16x32_bf16(false, a, false, b, (short)0, c,
                                                 false, false);
}

// ---------------------------------------------------------------------------
// fp32 -> bf16 (stored as ushort) elementwise convert
// ---------------------------------------------------------------------------
__global__ void k_f32_to_bf16(const float* __restrict__ in,
                              unsigned short* __restrict__ out, int n) {
  int i = blockIdx.x * blockDim.x + threadIdx.x;
  if (i < n) out[i] = f2bf(in[i]);
}

// ---------------------------------------------------------------------------
// Weight (E x E f32, row-major K x N) -> bf16 transposed W^T (N x K row-major).
// One-shot; makes every GEMM B-fragment a contiguous 32-byte run.
// ---------------------------------------------------------------------------
__global__ void k_transpose_w_bf16(const float* __restrict__ w,
                                   unsigned short* __restrict__ wT) {
  int idx = blockIdx.x * blockDim.x + threadIdx.x;   // E*E
  if (idx >= EMB * EMB) return;
  int k = idx & (EMB - 1);
  int n = idx >> 10;
  wT[idx] = f2bf(w[(size_t)k * EMB + n]);            // writes coalesced over k
}

// ---------------------------------------------------------------------------
// V: f32 (S, E) viewed as (S, NH, VD)  ->  bf16 vT (NH, VD, S)
// Makes attention P.V B-fragments contiguous.
// ---------------------------------------------------------------------------
__global__ void k_transpose_v(const float* __restrict__ vf,
                              unsigned short* __restrict__ vT) {
  int idx = blockIdx.x * blockDim.x + threadIdx.x;     // NH*VD*S = 2^21
  if (idx >= NH * VD * S_LEN) return;
  int s = idx & (S_LEN - 1);
  int d = (idx >> 11) & (VD - 1);
  int h = idx >> 18;
  vT[idx] = f2bf(vf[(size_t)s * EMB + h * VD + d]);
}

// ---------------------------------------------------------------------------
// C[M,N](f32) = A[M,K](bf16, row-major) @ BT[N,K](bf16, row-major = B^T).
// LDS-free: every WMMA operand is two aligned global_load_b128; the working
// set (A slab + weight) is L2-resident (192 MB L2), so intra-block B re-reads
// are cache-served. 128 threads = 4 waves; block tile 64x64; K slab 64
// (20 b128 loads batched, then 8 WMMAs).
// ---------------------------------------------------------------------------
__global__ __launch_bounds__(128)
void k_gemm_bf16(const unsigned short* __restrict__ A,
                 const unsigned short* __restrict__ BT,
                 float* __restrict__ C, int M, int N, int K) {
  const int t    = threadIdx.x;
  const int wave = t >> 5;
  const int lane = t & 31;
  const int lrow = lane & 15;
  const int hi   = lane >> 4;          // 0: lanes 0-15, 1: lanes 16-31
  const int blockRow = blockIdx.y * 64;
  const int blockCol = blockIdx.x * 64;

  const int g0 = hi * 16, g1 = hi * 16 + 8;  // A-frag K groups (contiguous k = 16*hi + s)
  const int klo = hi * 16;                   // B-frag K offset

  v8f acc[4] = {};

  const unsigned short* arow = A + (size_t)(blockRow + wave * 16 + lrow) * K;
  const unsigned short* brow[4];
#pragma unroll
  for (int tile = 0; tile < 4; tile++)
    brow[tile] = BT + (size_t)(blockCol + tile * 16 + lrow) * K;

  for (int k0 = 0; k0 < K; k0 += 64) {
    // ---- batch all fragment loads for this K slab (20x global_load_b128)
    BF16Frag af[2], bfr[2][4];
#pragma unroll
    for (int ch = 0; ch < 2; ch++) {
      af[ch].q[0] = *(const uint4*)&arow[k0 + ch * 32 + g0];
      af[ch].q[1] = *(const uint4*)&arow[k0 + ch * 32 + g1];
#pragma unroll
      for (int tile = 0; tile < 4; tile++) {
        const unsigned short* bp = &brow[tile][k0 + ch * 32 + klo];
        bfr[ch][tile].q[0] = *(const uint4*)&bp[0];
        bfr[ch][tile].q[1] = *(const uint4*)&bp[8];
      }
    }
    if (k0 + 64 < K) {   // prefetch next K slab (global_prefetch_b8)
      __builtin_prefetch(&arow[k0 + 64 + klo], 0, 3);
      __builtin_prefetch(&brow[lane & 3][k0 + 64 + klo], 0, 3);
    }
    // ---- 8 WMMAs on the slab
#pragma unroll
    for (int ch = 0; ch < 2; ch++)
#pragma unroll
      for (int tile = 0; tile < 4; tile++)
        acc[tile] = wmma_bf16(af[ch].v, bfr[ch][tile].v, acc[tile]);
  }

  // C layout: VGPR j -> row (j + hi*8), col = lane%16 within the 16x16 tile
#pragma unroll
  for (int tile = 0; tile < 4; tile++) {
#pragma unroll
    for (int j = 0; j < 8; j++) {
      int row = blockRow + wave * 16 + j + hi * 8;
      int col = blockCol + tile * 16 + lrow;
      C[(size_t)row * N + col] = acc[tile][j];
    }
  }
}

// ---------------------------------------------------------------------------
// Interleaved RoPE on (S, 16, 64) fp32 -> bf16, same layout.
// ---------------------------------------------------------------------------
__global__ void k_rope_bf16(const float* __restrict__ in,
                            const float* __restrict__ cosb,
                            const float* __restrict__ sinb,
                            unsigned short* __restrict__ out) {
  int idx = blockIdx.x * blockDim.x + threadIdx.x;   // S*16*32
  if (idx >= S_LEN * NQH * 32) return;
  int i  = idx & 31;
  int hh = (idx >> 5) & 15;
  int s  = idx >> 9;
  const float* row = in + ((size_t)s * NQH + hh) * HD;
  float x1 = row[2 * i], x2 = row[2 * i + 1];
  float c = cosb[s * 32 + i], sn = sinb[s * 32 + i];
  unsigned short* orow = out + ((size_t)s * NQH + hh) * HD;
  orow[2 * i]     = f2bf(x1 * c - x2 * sn);
  orow[2 * i + 1] = f2bf(x1 * sn + x2 * c);
}

// ---------------------------------------------------------------------------
// lambda_full = exp(sum lq1*lk1) - exp(sum lq2*lk2) + LAMBDA_INIT
// ---------------------------------------------------------------------------
__global__ void k_lambda(const float* __restrict__ lq1, const float* __restrict__ lk1,
                         const float* __restrict__ lq2, const float* __restrict__ lk2,
                         float* __restrict__ lam_out) {
  float s1 = 0.f, s2 = 0.f;
  for (int i = 0; i < HD; i++) { s1 += lq1[i] * lk1[i]; s2 += lq2[i] * lk2[i]; }
  *lam_out = __expf(s1) - __expf(s2) + LAMBDA_INIT_F;
}

// ---------------------------------------------------------------------------
// Fused differential flash attention + RMSNorm epilogue.
// Grid: (S/16, NH), one wave per block; both component attentions (heads 2h,
// 2h+1) streamed over 32-key blocks with online softmax; epilogue does
// a1 - lam*a2, RMSNorm(128), subln scale, (1-lambda_init).
// q/k: bf16 (S,16,64); vT: bf16 (NH,VD,S); attn out: f32 (S, 1024).
// ---------------------------------------------------------------------------
__global__ __launch_bounds__(32)
void k_diff_flash_attn(const unsigned short* __restrict__ qb,
                       const unsigned short* __restrict__ kb,
                       const unsigned short* __restrict__ vT,
                       const float* __restrict__ lam_p,
                       const float* __restrict__ subln,
                       float* __restrict__ attn) {
  __shared__ __align__(16) unsigned short sP[16][32];

  const int   qblk = blockIdx.x;
  const int   h    = blockIdx.y;
  const int   lane = threadIdx.x;
  const int   lrow = lane & 15;
  const int   hi   = lane >> 4;
  const int   g0 = hi * 16, g1 = hi * 16 + 8;
  const int   klo = hi * 16;
  const float scale = 0.125f;               // 1/sqrt(64)

  // Q fragments: [component][k-chunk], A-layout 16x32 (vectorized b128 loads)
  v16bf qf[2][2];
#pragma unroll
  for (int c = 0; c < 2; c++) {
#pragma unroll
    for (int ch = 0; ch < 2; ch++) {
      const int head = 2 * h + c;
      const unsigned short* qrow =
          qb + ((size_t)(qblk * 16 + lrow) * NQH + head) * HD + ch * 32;
      BF16Frag f;
      f.q[0] = *(const uint4*)&qrow[g0];
      f.q[1] = *(const uint4*)&qrow[g1];
      qf[c][ch] = f.v;
    }
  }

  float mrow[2][8], lsum[2][8];
  v8f   o[2][8] = {};
#pragma unroll
  for (int c = 0; c < 2; c++)
#pragma unroll
    for (int j = 0; j < 8; j++) { mrow[c][j] = -1e30f; lsum[c][j] = 0.f; }

  const int kmax = (qblk + 1) * 16;          // causal horizon (exclusive)

  for (int kb0 = 0; kb0 < kmax; kb0 += 32) {
    if (kb0 + 32 < kmax) {   // prefetch next key block (K and V^T lines)
      __builtin_prefetch(kb + ((size_t)(kb0 + 32 + lrow) * NQH + 2 * h) * HD, 0, 3);
      __builtin_prefetch(vT + ((size_t)(h * VD + lane * 4)) * S_LEN + kb0 + 32, 0, 3);
    }
#pragma unroll
    for (int c = 0; c < 2; c++) {
      const int head = 2 * h + c;

      // ---- scores: Q(16x64) . K^T -> two 16x16 tiles (keys kb0.. / kb0+16..)
      v8f sc[2] = {};
#pragma unroll
      for (int tl = 0; tl < 2; tl++) {
        const int key = kb0 + tl * 16 + lrow;          // <= 2047 by construction
#pragma unroll
        for (int ch = 0; ch < 2; ch++) {
          const unsigned short* krow =
              kb + ((size_t)key * NQH + head) * HD + ch * 32 + klo;
          BF16Frag bt;
          bt.q[0] = *(const uint4*)&krow[0];
          bt.q[1] = *(const uint4*)&krow[8];
          sc[tl] = wmma_bf16(qf[c][ch], bt.v, sc[tl]);
        }
      }

      __syncthreads();   // sP reuse guard (single wave: cheap)

      // ---- online softmax per row (rows j+hi*8 striped across 16 lanes)
#pragma unroll
      for (int j = 0; j < 8; j++) {
        const int m    = j + hi * 8;
        const int qidx = qblk * 16 + m;
        float s0 = sc[0][j] * scale;
        float s1 = sc[1][j] * scale;
        if (kb0 + lrow > qidx)      s0 = -1e30f;
        if (kb0 + 16 + lrow > qidx) s1 = -1e30f;
        float mx = fmaxf(s0, s1);
#pragma unroll
        for (int d = 1; d < 16; d <<= 1) mx = fmaxf(mx, __shfl_xor(mx, d, 32));
        const float mnew  = fmaxf(mrow[c][j], mx);
        const float alpha = __expf(mrow[c][j] - mnew);
        const float p0 = __expf(s0 - mnew);
        const float p1 = __expf(s1 - mnew);
        float ps = p0 + p1;
#pragma unroll
        for (int d = 1; d < 16; d <<= 1) ps += __shfl_xor(ps, d, 32);
        lsum[c][j] = lsum[c][j] * alpha + ps;
        mrow[c][j] = mnew;
#pragma unroll
        for (int t8 = 0; t8 < 8; t8++) o[c][t8][j] *= alpha;
        sP[m][lrow]      = f2bf(p0);
        sP[m][16 + lrow] = f2bf(p1);
      }

      __syncthreads();

      // ---- P(16x32) . V(32x128): A-frag from LDS (b128), B-frags from vT
      BF16Frag pf;
      pf.q[0] = *(const uint4*)&sP[lrow][g0];
      pf.q[1] = *(const uint4*)&sP[lrow][g1];

#pragma unroll
      for (int t8 = 0; t8 < 8; t8++) {
        BF16Frag vt;
        const unsigned short* vp =
            vT + ((size_t)(h * VD + t8 * 16 + lrow)) * S_LEN + kb0 + klo;
        vt.q[0] = *(const uint4*)&vp[0];
        vt.q[1] = *(const uint4*)&vp[8];
        o[c][t8] = wmma_bf16(pf.v, vt.v, o[c][t8]);
      }
    }
  }

  // ---- epilogue: a1 - lam*a2, RMSNorm(128), subln, (1 - lambda_init)
  const float lam  = *lam_p;
  const float gain = 1.0f - LAMBDA_INIT_F;
#pragma unroll
  for (int j = 0; j < 8; j++) {
    const int m    = j + hi * 8;
    const int qidx = qblk * 16 + m;
    const float inv1 = 1.0f / lsum[0][j];
    const float inv2 = 1.0f / lsum[1][j];
    float vals[8];
    float ss = 0.f;
#pragma unroll
    for (int t8 = 0; t8 < 8; t8++) {
      const float a1 = o[0][t8][j] * inv1;
      const float a2 = o[1][t8][j] * inv2;
      const float v  = a1 - lam * a2;
      vals[t8] = v;
      ss += v * v;
    }
#pragma unroll
    for (int d = 1; d < 16; d <<= 1) ss += __shfl_xor(ss, d, 32);
    const float r = rsqrtf(ss * (1.0f / (float)VD) + RMS_EPS_F);
#pragma unroll
    for (int t8 = 0; t8 < 8; t8++) {
      const int col = t8 * 16 + lrow;
      attn[(size_t)qidx * EMB + h * VD + col] = vals[t8] * r * subln[col] * gain;
    }
  }
}

// ---------------------------------------------------------------------------
// Host launcher
// ---------------------------------------------------------------------------
extern "C" void kernel_launch(void* const* d_in, const int* in_sizes, int n_in,
                              void* d_out, int out_size, void* d_ws, size_t ws_size,
                              hipStream_t stream) {
  (void)in_sizes; (void)n_in; (void)out_size; (void)ws_size;

  const float* x     = (const float*)d_in[0];
  const float* cosb  = (const float*)d_in[1];
  const float* sinb  = (const float*)d_in[2];
  const float* wq    = (const float*)d_in[3];
  const float* wk    = (const float*)d_in[4];
  const float* wv    = (const float*)d_in[5];
  const float* wo    = (const float*)d_in[6];
  const float* lq1   = (const float*)d_in[7];
  const float* lk1   = (const float*)d_in[8];
  const float* lq2   = (const float*)d_in[9];
  const float* lk2   = (const float*)d_in[10];
  const float* subln = (const float*)d_in[11];
  float* out = (float*)d_out;

  char* ws = (char*)d_ws;
  size_t off = 0;
  auto take = [&](size_t b) { size_t o = off; off += (b + 255) & ~(size_t)255; return o; };

  const size_t nX = (size_t)S_LEN * EMB;   // 2M elements
  const size_t nW = (size_t)EMB * EMB;     // 1M elements

  unsigned short* xb    = (unsigned short*)(ws + take(nX * 2));
  unsigned short* wqbT  = (unsigned short*)(ws + take(nW * 2));
  unsigned short* wkbT  = (unsigned short*)(ws + take(nW * 2));
  unsigned short* wvbT  = (unsigned short*)(ws + take(nW * 2));
  unsigned short* wobT  = (unsigned short*)(ws + take(nW * 2));
  float*          qf    = (float*)(ws + take(nX * 4));
  float*          kf    = (float*)(ws + take(nX * 4));
  float*          vf    = (float*)(ws + take(nX * 4));
  unsigned short* qbf   = (unsigned short*)(ws + take(nX * 2));
  unsigned short* kbf   = (unsigned short*)(ws + take(nX * 2));
  unsigned short* vT    = (unsigned short*)(ws + take(nX * 2));  // (NH, VD, S)
  float*          lamp  = (float*)(ws + take(256));
  // aliases (stream-ordered reuse; producers of qf/kf fully consumed first)
  float*          attnf  = qf;                    // f32 attention output (S, E)
  unsigned short* attnbf = (unsigned short*)kf;   // bf16 attention output

  const int CONV = 256;
  k_f32_to_bf16<<<(int)((nX + CONV - 1) / CONV), CONV, 0, stream>>>(x, xb, (int)nX);
  k_transpose_w_bf16<<<(int)((nW + CONV - 1) / CONV), CONV, 0, stream>>>(wq, wqbT);
  k_transpose_w_bf16<<<(int)((nW + CONV - 1) / CONV), CONV, 0, stream>>>(wk, wkbT);
  k_transpose_w_bf16<<<(int)((nW + CONV - 1) / CONV), CONV, 0, stream>>>(wv, wvbT);
  k_transpose_w_bf16<<<(int)((nW + CONV - 1) / CONV), CONV, 0, stream>>>(wo, wobT);

  dim3 ggrid(EMB / 64, S_LEN / 64);
  k_gemm_bf16<<<ggrid, 128, 0, stream>>>(xb, wqbT, qf, S_LEN, EMB, EMB);
  k_gemm_bf16<<<ggrid, 128, 0, stream>>>(xb, wkbT, kf, S_LEN, EMB, EMB);
  k_gemm_bf16<<<ggrid, 128, 0, stream>>>(xb, wvbT, vf, S_LEN, EMB, EMB);

  const int nR = S_LEN * NQH * 32;
  k_rope_bf16<<<(nR + CONV - 1) / CONV, CONV, 0, stream>>>(qf, cosb, sinb, qbf);
  k_rope_bf16<<<(nR + CONV - 1) / CONV, CONV, 0, stream>>>(kf, cosb, sinb, kbf);
  k_transpose_v<<<(int)((nX + CONV - 1) / CONV), CONV, 0, stream>>>(vf, vT);

  k_lambda<<<1, 1, 0, stream>>>(lq1, lk1, lq2, lk2, lamp);

  dim3 agrid(S_LEN / 16, NH);
  k_diff_flash_attn<<<agrid, 32, 0, stream>>>(qbf, kbf, vT, lamp, subln, attnf);

  k_f32_to_bf16<<<(int)((nX + CONV - 1) / CONV), CONV, 0, stream>>>(attnf, attnbf, (int)nX);
  k_gemm_bf16<<<ggrid, 128, 0, stream>>>(attnbf, wobT, out, S_LEN, EMB, EMB);
}
